// lossYoloV2_31241592111592
// MI455X (gfx1250) — compile-verified, matrix operand order checked by reading, not run on hardware
//
#include <hip/hip_runtime.h>

// ---- problem constants (match reference) ----
#define YNUM        5
#define ANCHOR_LEN  25        // CLASSES(20) + COORDS(4) + 1
#define BATCH       16
#define HH          96
#define WWW         96
#define HW          (HH * WWW)      // 9216
#define NT          30
#define THRESH      0.5f
#define EPS_W       1e-05f
#define BLK         256
#define BLKS_PER_BA (HW / BLK)      // 36
#define NPARTIAL    (BLKS_PER_BA * BATCH * YNUM)  // 2880

typedef __attribute__((ext_vector_type(16))) _Float16 v16h;
typedef __attribute__((ext_vector_type(8)))  float    v8f;

#if defined(__gfx1250__) && __has_builtin(__builtin_amdgcn_global_load_async_to_lds_b128) && __has_builtin(__builtin_amdgcn_s_wait_asynccnt)
#define USE_ASYNC_LDS 1
typedef int v4i __attribute__((vector_size(16)));           // matches builtin param
typedef __attribute__((address_space(1))) v4i gv4i;         // global
typedef __attribute__((address_space(3))) v4i lv4i;         // LDS
#endif

// Per-wave (wave32) sum of `acc` via the matrix pipe:
// A(16x32 f16): lane L puts acc into K-slot 0 (lanes 0-15: row L, K=0;
// lanes 16-31: row L-16, K=8). B = ones => D[m][n] = acc[m] + acc[m+16].
// Each lane sums its 8 f32 D-VGPRs, then one shfl_xor(16) gives the wave total.
__device__ __forceinline__ float wave_reduce_wmma(float acc) {
  v16h A = {};
  A[0] = (_Float16)acc;
  v16h B;
#pragma unroll
  for (int i = 0; i < 16; ++i) B[i] = (_Float16)1.0f;
  v8f C = {};
  C = __builtin_amdgcn_wmma_f32_16x16x32_f16(false, A, false, B,
                                             (short)0, C, false, false);
  float s = ((C[0] + C[1]) + (C[2] + C[3])) + ((C[4] + C[5]) + (C[6] + C[7]));
  s += __shfl_xor(s, 16, 32);
  return s;  // identical in all lanes
}

__global__ __launch_bounds__(BLK) void yolo_noobj_partial(
    const float* __restrict__ x, const float* __restrict__ truth,
    float* __restrict__ ws) {
  __shared__ __align__(16) float s_raw[NT * 4];
  __shared__ float s_tx1[NT], s_ty1[NT], s_tx2[NT], s_ty2[NT], s_ta[NT];
  __shared__ int   s_nvalid;
  __shared__ float s_wsum[BLK / 32];

  const int tid = threadIdx.x;
  const int ba  = blockIdx.y;            // 0..79 : (batch, anchor)
  const int b   = ba / YNUM;
  const int a   = ba % YNUM;
  const int hw  = blockIdx.x * BLK + tid;

  // ---- stage 30 truth rows (480 B) into LDS ----
  const float* trow = truth + (size_t)b * NT * 4;
#ifdef USE_ASYNC_LDS
  if (tid < NT) {
    __builtin_amdgcn_global_load_async_to_lds_b128(
        (gv4i*)(trow + tid * 4), (lv4i*)&s_raw[tid * 4], 0, 0);
  }
  __builtin_amdgcn_s_wait_asynccnt(0);
#else
  if (tid < NT) {
    const float4 r = *(const float4*)(trow + tid * 4);
    *(float4*)&s_raw[tid * 4] = r;
  }
#endif
  __syncthreads();

  if (tid < NT) {
    const float cx = s_raw[tid * 4 + 0], cy = s_raw[tid * 4 + 1];
    const float w  = s_raw[tid * 4 + 2], h  = s_raw[tid * 4 + 3];
    s_tx1[tid] = cx - 0.5f * w;  s_ty1[tid] = cy - 0.5f * h;
    s_tx2[tid] = cx + 0.5f * w;  s_ty2[tid] = cy + 0.5f * h;
    s_ta[tid]  = w * h;
  }
  if (tid == 0) {
    // cumprod validity: first truth with w < EPS invalidates the rest
    int nv = NT;
    for (int t = 0; t < NT; ++t)
      if (!(s_raw[t * 4 + 2] >= EPS_W)) { nv = t; break; }
    s_nvalid = nv;
  }
  __syncthreads();

  // ---- load this box's 5 plane values (coalesced across hw) ----
  const size_t base = ((size_t)(b * (YNUM * ANCHOR_LEN) + a * ANCHOR_LEN)) * HW + hw;
  const float px  = x[base];
  const float py  = x[base + HW];
  const float pw  = x[base + 2 * (size_t)HW];
  const float ph  = x[base + 3 * (size_t)HW];
  const float obj = x[base + 4 * (size_t)HW];

  const float px1 = px - 0.5f * pw, px2 = px + 0.5f * pw;
  const float py1 = py - 0.5f * ph, py2 = py + 0.5f * ph;
  const float ap  = pw * ph;

  // best_iou < 0.5  <=>  every valid truth has iou < 0.5.
  // iou >= 0.5 (with union > 0) <=> inter >= 0.5 * union  -> no divide needed.
  const int nv = s_nvalid;
  bool noobj = true;
  for (int t = 0; t < nv; ++t) {
    const float iw = fminf(px2, s_tx2[t]) - fmaxf(px1, s_tx1[t]);
    const float ih = fminf(py2, s_ty2[t]) - fmaxf(py1, s_ty1[t]);
    if (iw >= 0.0f && ih >= 0.0f) {
      const float inter = iw * ih;
      const float uni   = ap + s_ta[t] - inter;
      if (uni > 0.0f && inter >= THRESH * uni) { noobj = false; break; }
    }
  }
  const float acc = noobj ? obj * obj : 0.0f;

  // ---- block reduction: wave32 WMMA reduce, then 8 wave sums ----
  const float wsum = wave_reduce_wmma(acc);   // fully converged here (EXEC all 1s)
  if ((tid & 31) == 0) s_wsum[tid >> 5] = wsum;
  __syncthreads();
  if (tid == 0) {
    float bs = 0.0f;
#pragma unroll
    for (int i = 0; i < BLK / 32; ++i) bs += s_wsum[i];
    ws[blockIdx.y * BLKS_PER_BA + blockIdx.x] = bs;
  }
}

__global__ __launch_bounds__(BLK) void yolo_noobj_final(
    const float* __restrict__ ws, float* __restrict__ out, int n) {
  __shared__ float sm[BLK];
  float s = 0.0f;
  for (int i = threadIdx.x; i < n; i += BLK) s += ws[i];  // fixed order
  sm[threadIdx.x] = s;
  __syncthreads();
  for (int off = BLK / 2; off > 0; off >>= 1) {
    if ((int)threadIdx.x < off) sm[threadIdx.x] += sm[threadIdx.x + off];
    __syncthreads();
  }
  if (threadIdx.x == 0) out[0] = sm[0];
}

extern "C" void kernel_launch(void* const* d_in, const int* in_sizes, int n_in,
                              void* d_out, int out_size, void* d_ws, size_t ws_size,
                              hipStream_t stream) {
  (void)in_sizes; (void)n_in; (void)out_size; (void)ws_size;
  const float* x     = (const float*)d_in[0];   // (16, 125, 96, 96) f32
  const float* truth = (const float*)d_in[1];   // (16, 30, 4) f32
  float* out = (float*)d_out;                   // scalar f32
  float* ws  = (float*)d_ws;                    // 2880 partial sums

  dim3 grid(BLKS_PER_BA, BATCH * YNUM);         // (36, 80) blocks of 256
  yolo_noobj_partial<<<grid, BLK, 0, stream>>>(x, truth, ws);
  yolo_noobj_final<<<1, BLK, 0, stream>>>(ws, out, NPARTIAL);
}
